// FusionNetwork_two_fusion_6055903887368
// MI455X (gfx1250) — matrix operand order repaired and checked by reading, not compiled, not run on hardware
//
#include <hip/hip_runtime.h>
#include <hip/hip_bf16.h>
#include <math.h>

// ---------------------------------------------------------------------------
// MI455X (gfx1250) implementation.
// GEMMs run as bf16 WMMA (v_wmma_f32_16x16x32_bf16, f32 accumulate).
// Block tile 128x256, 8 waves, each wave computes 64x64 (4x4 fragments,
// 16 v_wmma per BK=32 step). Everything else is bandwidth-bound float4 work.
// ---------------------------------------------------------------------------

typedef __bf16 bf16;
typedef float v8f  __attribute__((ext_vector_type(8)));
typedef bf16  v16bf __attribute__((ext_vector_type(16)));

#define KWIN  11
#define BATCH 1024
#define DSEQ  1280
#define DSTR  512
#define EMB1  1792          // D_SEQ + D_STR
#define EMB2  2564          // 2*D_SEQ + 4
#define LDA2  2592          // EMB2 padded to multiple of 32 for GEMM lda
#define MROWS (BATCH * KWIN) // 11264 = 88 * 128 (always multiple of 128)
#define NCHUNK 44           // row chunks for BN partial sums (11264/44 = 256)

// Residue property bitmask: bit3=aliphatic, bit2=aromatic, bit1=acidic, bit0=basic
// order: G A S P V T C I L N D Q K E M H F R Y W X
__constant__ unsigned char RESP[21] = {
    0, 8, 0, 0, 8, 0, 0, 8, 8, 0, 2, 0, 1, 2, 8, 1, 4, 1, 4, 4, 0};

// ---------------------------------------------------------------------------
// kin = mean over rows 1..127 of node_output_kin  (B,128,DSEQ) -> (B,DSEQ)
// ---------------------------------------------------------------------------
__global__ void kin_mean_kernel(const float* __restrict__ kin_in,
                                float* __restrict__ kin_out) {
    int d4 = blockIdx.x * blockDim.x + threadIdx.x;   // DSEQ/4 = 320 per b
    int b = blockIdx.y;
    if (d4 >= DSEQ / 4) return;
    const float4* p = (const float4*)(kin_in + ((size_t)b * 128 + 1) * DSEQ) + d4;
    float4 s = {0.f, 0.f, 0.f, 0.f};
    for (int l = 0; l < 127; ++l) {
        float4 v = p[(size_t)l * (DSEQ / 4)];
        s.x += v.x; s.y += v.y; s.z += v.z; s.w += v.w;
    }
    const float r = 1.0f / 127.0f;
    float4 o = {s.x * r, s.y * r, s.z * r, s.w * r};
    ((float4*)(kin_out + (size_t)b * DSEQ))[d4] = o;
}

// ---------------------------------------------------------------------------
// Build flat (MROWS x EMB1): [gf1 | gf2] + phos_emb at k==5.
// Also writes node_feature (= gf1) into d_out.  float4 vectorized.
// ---------------------------------------------------------------------------
__global__ void build_flat_kernel(const float* __restrict__ sub,
                                  const float* __restrict__ no2,
                                  const float* __restrict__ phos,
                                  const int* __restrict__ pos,
                                  float* __restrict__ flat,
                                  float* __restrict__ nodefeat) {
    int row = blockIdx.x;
    int b = row / KWIN, k = row - b * KWIN;
    int src2 = pos[b] + k - 5;                 // pad K//2 = 5 each side
    bool in2 = (src2 >= 0) && (src2 < 128);
    const float4* g1 = (const float4*)(sub + ((size_t)b * 64 + (k + 1)) * DSEQ);
    const float4* g2 = (const float4*)(no2 + ((size_t)b * 128 + (in2 ? src2 : 0)) * DSTR);
    const float4* ph = (const float4*)phos;
    float4* out = (float4*)(flat + (size_t)row * EMB1);
    float4* nf = (float4*)(nodefeat + (size_t)row * DSEQ);
    for (int c4 = threadIdx.x; c4 < EMB1 / 4; c4 += blockDim.x) {
        float4 v;
        if (c4 < DSEQ / 4) {
            v = g1[c4];
            nf[c4] = v;
        } else {
            float4 z = {0.f, 0.f, 0.f, 0.f};
            v = in2 ? g2[c4 - DSEQ / 4] : z;
        }
        if (k == 5) {
            float4 p4 = ph[c4];
            v.x += p4.x; v.y += p4.y; v.z += p4.z; v.w += p4.w;
        }
        out[c4] = v;
    }
}

// ---------------------------------------------------------------------------
// BatchNorm stats, two-phase (partial row-chunk sums -> finalize).
// ---------------------------------------------------------------------------
__global__ void bn_partial_kernel(const float* __restrict__ X, int rows, int F,
                                  int ld, float* __restrict__ psum,
                                  float* __restrict__ psq) {
    int f = blockIdx.x * blockDim.x + threadIdx.x;
    int chunk = blockIdx.y;
    if (f >= F) return;
    int per = (rows + NCHUNK - 1) / NCHUNK;
    int r0 = chunk * per;
    int r1 = min(rows, r0 + per);
    float s = 0.f, sq = 0.f;
    const float* p = X + f;
    for (int r = r0; r < r1; ++r) {
        float v = p[(size_t)r * ld];
        s += v;
        sq += v * v;
    }
    psum[(size_t)chunk * F + f] = s;
    psq[(size_t)chunk * F + f] = sq;
}

__global__ void bn_finalize_kernel(int rows, int F,
                                   const float* __restrict__ psum,
                                   const float* __restrict__ psq,
                                   float* __restrict__ mean,
                                   float* __restrict__ rstd) {
    int f = blockIdx.x * blockDim.x + threadIdx.x;
    if (f >= F) return;
    float s = 0.f, sq = 0.f;
    for (int c = 0; c < NCHUNK; ++c) {
        s += psum[(size_t)c * F + f];
        sq += psq[(size_t)c * F + f];
    }
    float m = s / rows;
    float var = sq / rows - m * m;
    mean[f] = m;
    rstd[f] = rsqrtf(var + 1e-5f);
}

// ---------------------------------------------------------------------------
// Two BN+ReLU activations (different gamma/beta, same stats) -> bf16.
// ---------------------------------------------------------------------------
__global__ void act2_kernel(const float* __restrict__ X,
                            const float* __restrict__ mean,
                            const float* __restrict__ rstd,
                            const float* __restrict__ ga, const float* __restrict__ ba,
                            const float* __restrict__ gb, const float* __restrict__ bb,
                            bf16* __restrict__ A1, bf16* __restrict__ A2,
                            int F, int ldin, int ldout) {
    int row = blockIdx.x;
    const float4* src = (const float4*)(X + (size_t)row * ldin);
    bf16* o1 = A1 + (size_t)row * ldout;
    bf16* o2 = A2 + (size_t)row * ldout;
    for (int q = threadIdx.x; q < F / 4; q += blockDim.x) {
        float4 x = src[q];
        float4 m = ((const float4*)mean)[q];
        float4 rs = ((const float4*)rstd)[q];
        float4 g1 = ((const float4*)ga)[q], b1 = ((const float4*)ba)[q];
        float4 g2 = ((const float4*)gb)[q], b2 = ((const float4*)bb)[q];
        float xn0 = (x.x - m.x) * rs.x, xn1 = (x.y - m.y) * rs.y;
        float xn2 = (x.z - m.z) * rs.z, xn3 = (x.w - m.w) * rs.w;
        union { uint2 u; bf16 h[4]; } p1, p2;
        float v;
        v = xn0 * g1.x + b1.x; p1.h[0] = (bf16)(v > 0.f ? v : 0.f);
        v = xn1 * g1.y + b1.y; p1.h[1] = (bf16)(v > 0.f ? v : 0.f);
        v = xn2 * g1.z + b1.z; p1.h[2] = (bf16)(v > 0.f ? v : 0.f);
        v = xn3 * g1.w + b1.w; p1.h[3] = (bf16)(v > 0.f ? v : 0.f);
        v = xn0 * g2.x + b2.x; p2.h[0] = (bf16)(v > 0.f ? v : 0.f);
        v = xn1 * g2.y + b2.y; p2.h[1] = (bf16)(v > 0.f ? v : 0.f);
        v = xn2 * g2.z + b2.z; p2.h[2] = (bf16)(v > 0.f ? v : 0.f);
        v = xn3 * g2.w + b2.w; p2.h[3] = (bf16)(v > 0.f ? v : 0.f);
        *(uint2*)(o1 + 4 * q) = p1.u;
        *(uint2*)(o2 + 4 * q) = p2.u;
    }
}

// ---------------------------------------------------------------------------
// WMMA GEMM: C[M,N] = A[M,K](bf16) * Bw[K,N](fp32 -> bf16 at tile fill) + bias
// Requirements: M % 128 == 0 (true here: M = 11264).
// Block tile 128x256, BK=32, 256 threads (8 waves). Wave grid 2(M) x 4(N):
// each wave computes 64x64 via 4x4 fragments -> 16 v_wmma per K step from
// 8 fragment loads (16 ds_load_b128).
// LDS is pre-swizzled to the CDNA5 fragment lane layout:
//   A (16x32, MxK):  lane = m%16 + 16*((k/8)%2),  idx = k%8 + 8*(k/16)
//   B (32x16, KxN):  lane = n%16 + 16*((k/8)%2),  idx = k%8 + 8*(k/16)
// Fill fast/slow selection is a wave-uniform scalar branch on the K-tile
// (only the single ragged tail tile of K=2564 takes the guarded path).
// mode 0: store fp32 acc+bias.  mode 1: store bf16 relu(acc+bias).
// ---------------------------------------------------------------------------
__global__ __launch_bounds__(256) void wmma_gemm_kernel(
    const bf16* __restrict__ A, int lda,
    const float* __restrict__ Bw, int ldb,
    const float* __restrict__ bias,
    int M, int N, int K,
    float* __restrict__ Cf, bf16* __restrict__ Cb, int ldc,
    int relu_bf16_out) {
    __shared__ __align__(32) bf16 Asm[2][8][32][16];   // [buf][mfrag][lane][idx] 16KB
    __shared__ __align__(32) bf16 Bsm[2][16][32][16];  // [buf][nfrag][lane][idx] 32KB

    const int t = threadIdx.x;
    const int m0 = blockIdx.y * 128;
    const int n0 = blockIdx.x * 256;
    const int KT = (K + 31) / 32;

    // ---- A tile fill: 2 threads per row; 32B global read -> 2x b128 LDS store
    auto fillA = [&](int buf, int kt, bool full) {
        int mloc = t >> 1;          // 0..127
        int khalf = t & 1;          // 16-wide K half of the 32-tile
        int mf = mloc >> 4, mrow = mloc & 15;
        int kb = kt * 32 + khalf * 16;
        const bf16* src = A + (size_t)(m0 + mloc) * lda + kb;
        if (full) {
            uint4 lo = *(const uint4*)(src);
            uint4 hi = *(const uint4*)(src + 8);
            *(uint4*)&Asm[buf][mf][mrow][8 * khalf] = lo;
            *(uint4*)&Asm[buf][mf][mrow + 16][8 * khalf] = hi;
        } else {
#pragma unroll
            for (int j = 0; j < 16; ++j) {
                bf16 v = (bf16)0.0f;
                if ((kb + j) < K) v = src[j];
                int lanesel = mrow + (((unsigned)j >> 3) & 1) * 16;
                int idx = (j & 7) + 8 * khalf;
                Asm[buf][mf][lanesel][idx] = v;
            }
        }
    };

    // ---- B tile fill: 1 thread per column (coalesced across threads),
    // 32 strided fp32 reads -> 16 packed b32 LDS stores.
    auto fillB = [&](int buf, int kt, bool full) {
        int nf = t >> 4, lanen = t & 15;
        int gn = n0 + t;
        const float* bp = Bw + (size_t)(kt * 32) * ldb + gn;
        if (full) {
            if (gn < N) {
#pragma unroll
                for (int kp = 0; kp < 16; ++kp) {
                    float b0 = bp[(size_t)(2 * kp) * ldb];
                    float b1 = bp[(size_t)(2 * kp + 1) * ldb];
                    union { unsigned u; bf16 h[2]; } pk;
                    pk.h[0] = (bf16)b0;
                    pk.h[1] = (bf16)b1;
                    *(unsigned*)&Bsm[buf][nf][lanen + 16 * ((kp >> 2) & 1)][2 * (kp & 3) + 8 * (kp >> 3)] = pk.u;
                }
                // prefetch next K tile of this weight column (global_prefetch_b8)
                if (kt * 32 + 64 < K) __builtin_prefetch(bp + (size_t)32 * ldb, 0, 1);
            } else {
#pragma unroll
                for (int kp = 0; kp < 16; ++kp)
                    *(unsigned*)&Bsm[buf][nf][lanen + 16 * (kp >> 3)][2 * (kp & 7)] = 0u;
            }
        } else {
#pragma unroll
            for (int kp = 0; kp < 16; ++kp) {
                int k = kt * 32 + 2 * kp;
                float b0 = (gn < N && k < K) ? Bw[(size_t)k * ldb + gn] : 0.f;
                float b1 = (gn < N && (k + 1) < K) ? Bw[(size_t)(k + 1) * ldb + gn] : 0.f;
                union { unsigned u; bf16 h[2]; } pk;
                pk.h[0] = (bf16)b0;
                pk.h[1] = (bf16)b1;
                *(unsigned*)&Bsm[buf][nf][lanen + 16 * ((kp >> 2) & 1)][2 * (kp & 3) + 8 * (kp >> 3)] = pk.u;
            }
        }
    };

    v8f acc[4][4] = {};
    const int wave = t >> 5, lane31 = t & 31;
    const int wm = wave & 1;   // M half of block tile (64 rows)
    const int wn = wave >> 1;  // N quarter of block tile (64 cols)

    {
        bool full0 = (32 <= K);
        fillA(0, 0, full0);
        fillB(0, 0, full0);
    }
    for (int kt = 0; kt < KT; ++kt) {
        __syncthreads();
        int cur = kt & 1;
        if (kt + 1 < KT) {
            bool full = (32 * (kt + 2) <= K);   // uniform scalar branch
            fillA(cur ^ 1, kt + 1, full);
            fillB(cur ^ 1, kt + 1, full);
        }
        v16bf af[4], bfr[4];
#pragma unroll
        for (int r = 0; r < 4; ++r)
            af[r] = *(const v16bf*)(&Asm[cur][wm * 4 + r][lane31][0]);
#pragma unroll
        for (int c = 0; c < 4; ++c)
            bfr[c] = *(const v16bf*)(&Bsm[cur][wn * 4 + c][lane31][0]);
#pragma unroll
        for (int r = 0; r < 4; ++r)
#pragma unroll
            for (int c = 0; c < 4; ++c)
                acc[r][c] = __builtin_amdgcn_wmma_f32_16x16x32_bf16(
                    false, af[r], false, bfr[c], (short)0, acc[r][c], false, false);
    }

    // Epilogue: C/D layout — lane 0-15: N=lane, M=v; lane 16-31: N=lane-16, M=v+8.
    const int ncol = lane31 & 15;
    const int mhalf = lane31 >> 4;
#pragma unroll
    for (int r = 0; r < 4; ++r) {
#pragma unroll
        for (int c = 0; c < 4; ++c) {
            int gn = n0 + wn * 64 + c * 16 + ncol;
            if (gn >= N) continue;
            float bb = bias[gn];
            int gmb = m0 + wm * 64 + r * 16 + mhalf * 8;
#pragma unroll
            for (int v = 0; v < 8; ++v) {
                int gm = gmb + v;
                float val = acc[r][c][v] + bb;
                if (relu_bf16_out) {
                    val = val > 0.f ? val : 0.f;
                    Cb[(size_t)gm * ldc + gn] = (bf16)val;
                } else {
                    Cf[(size_t)gm * ldc + gn] = val;
                }
            }
        }
    }
}

// ---------------------------------------------------------------------------
// gfeatN = sigmoid(F1)*gf1*a0 + F2*a1 ; also builds flat2 = [gfeatN|kin|props]
// ---------------------------------------------------------------------------
__global__ void combine1_kernel(const float* __restrict__ F1,
                                const float* __restrict__ F2,
                                const float* __restrict__ sub,
                                const float* __restrict__ kin,
                                const int* __restrict__ rawid,
                                const float* __restrict__ avec,
                                float* __restrict__ gfeatN,
                                float* __restrict__ flat2) {
    int row = blockIdx.x;
    int b = row / KWIN, k = row - b * KWIN;
    float a0 = avec[0], a1 = avec[1];
    const float4* f1p = (const float4*)(F1 + (size_t)row * DSEQ);
    const float4* f2p = (const float4*)(F2 + (size_t)row * DSEQ);
    const float4* g1p = (const float4*)(sub + ((size_t)b * 64 + (k + 1)) * DSEQ);
    const float4* kp = (const float4*)(kin + (size_t)b * DSEQ);
    float4* gp = (float4*)(gfeatN + (size_t)row * DSEQ);
    float4* fl = (float4*)(flat2 + (size_t)row * LDA2);
    for (int d4 = threadIdx.x; d4 < DSEQ / 4; d4 += blockDim.x) {
        float4 f1 = f1p[d4], f2 = f2p[d4], g1 = g1p[d4];
        float4 g;
        g.x = (1.f / (1.f + __expf(-f1.x))) * g1.x * a0 + f2.x * a1;
        g.y = (1.f / (1.f + __expf(-f1.y))) * g1.y * a0 + f2.y * a1;
        g.z = (1.f / (1.f + __expf(-f1.z))) * g1.z * a0 + f2.z * a1;
        g.w = (1.f / (1.f + __expf(-f1.w))) * g1.w * a0 + f2.w * a1;
        gp[d4] = g;
        fl[d4] = g;
        fl[DSEQ / 4 + d4] = kp[d4];
    }
    if (threadIdx.x < 4) {
        int id = rawid[b * KWIN + k];
        unsigned char m = RESP[id];
        float pv = ((m >> (3 - threadIdx.x)) & 1) ? 1.f : 0.f;
        flat2[(size_t)row * LDA2 + 2 * DSEQ + threadIdx.x] = pv;
    }
}

// ---------------------------------------------------------------------------
// graph_feature[b,d] = sum_k sigmoid(G1)*gfeatN*a2_0 + G2*a2_1
// ---------------------------------------------------------------------------
__global__ void final_kernel(const float* __restrict__ G1,
                             const float* __restrict__ G2,
                             const float* __restrict__ gfeatN,
                             const float* __restrict__ a2vec,
                             float* __restrict__ out) {
    int d4 = blockIdx.x * blockDim.x + threadIdx.x;
    int b = blockIdx.y;
    if (d4 >= DSEQ / 4) return;
    float a0 = a2vec[0], a1 = a2vec[1];
    float4 s = {0.f, 0.f, 0.f, 0.f};
    for (int k = 0; k < KWIN; ++k) {
        size_t roff = (size_t)(b * KWIN + k) * (DSEQ / 4) + d4;
        float4 f1 = ((const float4*)G1)[roff];
        float4 f2 = ((const float4*)G2)[roff];
        float4 g = ((const float4*)gfeatN)[roff];
        s.x += (1.f / (1.f + __expf(-f1.x))) * g.x * a0 + f2.x * a1;
        s.y += (1.f / (1.f + __expf(-f1.y))) * g.y * a0 + f2.y * a1;
        s.z += (1.f / (1.f + __expf(-f1.z))) * g.z * a0 + f2.z * a1;
        s.w += (1.f / (1.f + __expf(-f1.w))) * g.w * a0 + f2.w * a1;
    }
    ((float4*)(out + (size_t)b * DSEQ))[d4] = s;
}

// ---------------------------------------------------------------------------
extern "C" void kernel_launch(void* const* d_in, const int* in_sizes, int n_in,
                              void* d_out, int out_size, void* d_ws, size_t ws_size,
                              hipStream_t stream) {
    (void)in_sizes; (void)n_in; (void)out_size; (void)ws_size;

    const float* kin_in = (const float*)d_in[0];
    const float* sub    = (const float*)d_in[1];
    const float* no2    = (const float*)d_in[2];
    const float* avec   = (const float*)d_in[4];
    const float* a2vec  = (const float*)d_in[5];
    const float* phos   = (const float*)d_in[6];
    const float* g1g = (const float*)d_in[7],  *g1b  = (const float*)d_in[8];
    const float* g1W = (const float*)d_in[9],  *g1bs = (const float*)d_in[10];
    const float* r1g = (const float*)d_in[11], *r1b  = (const float*)d_in[12];
    const float* r1W1= (const float*)d_in[13], *r1b1 = (const float*)d_in[14];
    const float* r1W2= (const float*)d_in[15], *r1b2 = (const float*)d_in[16];
    const float* g2g = (const float*)d_in[17], *g2b  = (const float*)d_in[18];
    const float* g2W = (const float*)d_in[19], *g2bs = (const float*)d_in[20];
    const float* r2g = (const float*)d_in[21], *r2b  = (const float*)d_in[22];
    const float* r2W1= (const float*)d_in[23], *r2b1 = (const float*)d_in[24];
    const float* r2W2= (const float*)d_in[25], *r2b2 = (const float*)d_in[26];
    const int* pos   = (const int*)d_in[27];
    const int* rawid = (const int*)d_in[28];

    float* gout = (float*)d_out;                       // graph_feature (B,DSEQ)
    float* nodefeat = gout + (size_t)BATCH * DSEQ;     // node_feature (B,K,DSEQ)

    // ---- workspace layout (with reuse; stream order serializes hazards) ----
    char* ws = (char*)d_ws;
    size_t off = 0;
    auto alloc = [&](size_t bytes) -> void* {
        void* p = ws + off;
        off += (bytes + 255) & ~(size_t)255;
        return p;
    };
    float* kin   = (float*)alloc((size_t)BATCH * DSEQ * 4);
    float* mean1 = (float*)alloc((size_t)EMB1 * 4);
    float* rstd1 = (float*)alloc((size_t)EMB1 * 4);
    float* mean2 = (float*)alloc((size_t)LDA2 * 4);
    float* rstd2 = (float*)alloc((size_t)LDA2 * 4);
    float* psum  = (float*)alloc((size_t)NCHUNK * EMB2 * 4);
    float* psq   = (float*)alloc((size_t)NCHUNK * EMB2 * 4);
    float* flat  = (float*)alloc((size_t)MROWS * EMB1 * 4);  // reused as F1
    bf16*  X1    = (bf16*)alloc((size_t)MROWS * EMB1 * 2);
    bf16*  X2    = (bf16*)alloc((size_t)MROWS * EMB1 * 2);
    bf16*  X3    = (bf16*)alloc((size_t)MROWS * EMB1 * 2);
    float* gfeatN= (float*)alloc((size_t)MROWS * DSEQ * 4);
    float* flat2 = (float*)alloc((size_t)MROWS * LDA2 * 4);  // reused as G1|G2
    bf16*  X4    = (bf16*)alloc((size_t)MROWS * LDA2 * 2);   // reused as X6
    bf16*  X5    = (bf16*)alloc((size_t)MROWS * LDA2 * 2);

    float* F1 = flat;                                   // flat dead after act2
    float* F2 = (float*)X1;                             // X1(+X2) dead by then
    float* G1 = flat2;                                  // flat2 dead after act2#2
    float* G2 = flat2 + (size_t)MROWS * DSEQ;
    bf16*  X6 = X4;                                     // X4 dead after G1 gemm

    dim3 b256(256);

    // Stage 0: kin mean + feature assembly (+ node_feature output)
    kin_mean_kernel<<<dim3((DSEQ / 4 + 255) / 256, BATCH), b256, 0, stream>>>(kin_in, kin);
    build_flat_kernel<<<dim3(MROWS), b256, 0, stream>>>(sub, no2, phos, pos, flat, nodefeat);

    // Stage 1: BN stats, dual activation, three WMMA GEMMs
    bn_partial_kernel<<<dim3((EMB1 + 255) / 256, NCHUNK), b256, 0, stream>>>(
        flat, MROWS, EMB1, EMB1, psum, psq);
    bn_finalize_kernel<<<dim3((EMB1 + 255) / 256), b256, 0, stream>>>(
        MROWS, EMB1, psum, psq, mean1, rstd1);
    act2_kernel<<<dim3(MROWS), b256, 0, stream>>>(flat, mean1, rstd1, g1g, g1b, r1g, r1b,
                                                  X1, X2, EMB1, EMB1, EMB1);

    wmma_gemm_kernel<<<dim3(DSEQ / 256, MROWS / 128), b256, 0, stream>>>(
        X1, EMB1, g1W, DSEQ, g1bs, MROWS, DSEQ, EMB1, F1, nullptr, DSEQ, 0);
    wmma_gemm_kernel<<<dim3(EMB1 / 256, MROWS / 128), b256, 0, stream>>>(
        X2, EMB1, r1W1, EMB1, r1b1, MROWS, EMB1, EMB1, nullptr, X3, EMB1, 1);
    wmma_gemm_kernel<<<dim3(DSEQ / 256, MROWS / 128), b256, 0, stream>>>(
        X3, EMB1, r1W2, DSEQ, r1b2, MROWS, DSEQ, EMB1, F2, nullptr, DSEQ, 0);

    // Stage 2: gate + concat, BN stats, dual activation, three WMMA GEMMs
    combine1_kernel<<<dim3(MROWS), b256, 0, stream>>>(F1, F2, sub, kin, rawid, avec,
                                                      gfeatN, flat2);
    bn_partial_kernel<<<dim3((EMB2 + 255) / 256, NCHUNK), b256, 0, stream>>>(
        flat2, MROWS, EMB2, LDA2, psum, psq);
    bn_finalize_kernel<<<dim3((EMB2 + 255) / 256), b256, 0, stream>>>(
        MROWS, EMB2, psum, psq, mean2, rstd2);
    act2_kernel<<<dim3(MROWS), b256, 0, stream>>>(flat2, mean2, rstd2, g2g, g2b, r2g, r2b,
                                                  X4, X5, EMB2, LDA2, LDA2);

    wmma_gemm_kernel<<<dim3(DSEQ / 256, MROWS / 128), b256, 0, stream>>>(
        X4, LDA2, g2W, DSEQ, g2bs, MROWS, DSEQ, EMB2, G1, nullptr, DSEQ, 0);
    wmma_gemm_kernel<<<dim3((EMB2 + 255) / 256, MROWS / 128), b256, 0, stream>>>(
        X5, LDA2, r2W1, EMB2, r2b1, MROWS, EMB2, EMB2, nullptr, X6, LDA2, 1);
    wmma_gemm_kernel<<<dim3(DSEQ / 256, MROWS / 128), b256, 0, stream>>>(
        X6, LDA2, r2W2, DSEQ, r2b2, MROWS, DSEQ, EMB2, G2, nullptr, DSEQ, 0);

    // Final: gate, sum over K -> graph_feature
    final_kernel<<<dim3((DSEQ / 4 + 255) / 256, BATCH), b256, 0, stream>>>(
        G1, G2, gfeatN, a2vec, gout);
}
